// BitLinearStandard_9113920602308
// MI455X (gfx1250) — 8-bit matrix operand order repaired and checked by reading, compiled, not run on hardware
//
#include <hip/hip_runtime.h>
#include <math.h>

#define NS   16
#define C    256
#define H    64
#define W    64
#define HW   (H*W)
#define CHW  (C*HW)
#define KTOT (C*9)
#define WELEMS (C*C*9)

#define QB      128.0f
#define EPS     1e-6f
#define GN_EPS  1e-5f
#define WSCALE  0.01f

#define XQ_OFF 0
#define XQ_BYTES ((size_t)NS*CHW)
#define WQ_OFF XQ_BYTES
#define WQ_BYTES ((size_t)WELEMS)
#define ST_OFF (WQ_OFF + WQ_BYTES)

typedef __attribute__((ext_vector_type(8))) int v8i;
typedef int v4i_t __attribute__((vector_size(16)));
typedef __attribute__((address_space(1))) v4i_t* gv4p;
typedef __attribute__((address_space(3))) v4i_t* lv4p;

#if __has_builtin(__builtin_amdgcn_global_load_async_to_lds_b128)
#define USE_ASYNC 1
#else
#define USE_ASYNC 0
#endif

__device__ __forceinline__ void wait_async0() {
#if __has_builtin(__builtin_amdgcn_s_wait_asynccnt)
    __builtin_amdgcn_s_wait_asynccnt(0);
#else
    asm volatile("s_wait_asynccnt 0" ::: "memory");
#endif
}


__global__ void k_init(float* st) {
    if (threadIdx.x < 34) st[threadIdx.x] = 0.0f;
}

__global__ void k_gnstats(const float* __restrict__ x, float* __restrict__ st) {
    int s = blockIdx.x >> 6, chunk = blockIdx.x & 63;
    const float4* xs = (const float4*)(x + (size_t)s*CHW + (size_t)chunk*16384);
    float sum = 0.f, sq = 0.f;
    for (int i = threadIdx.x; i < 4096; i += 256) {
        float4 v = xs[i];
        sum += v.x + v.y + v.z + v.w;
        sq  += v.x*v.x + v.y*v.y + v.z*v.z + v.w*v.w;
    }
    __shared__ float r1[256], r2[256];
    r1[threadIdx.x] = sum; r2[threadIdx.x] = sq; __syncthreads();
    for (int o = 128; o > 0; o >>= 1) {
        if (threadIdx.x < o) { r1[threadIdx.x] += r1[threadIdx.x+o]; r2[threadIdx.x] += r2[threadIdx.x+o]; }
        __syncthreads();
    }
    if (threadIdx.x == 0) { atomicAdd(&st[s], r1[0]); atomicAdd(&st[16+s], r2[0]); }
}

__global__ void k_wabs(const float* __restrict__ w, float* __restrict__ st) {
    float s = 0.f;
    for (int i = blockIdx.x*256 + threadIdx.x; i < WELEMS; i += 576*256) s += fabsf(w[i]);
    __shared__ float r[256];
    r[threadIdx.x] = s; __syncthreads();
    for (int o = 128; o > 0; o >>= 1) {
        if (threadIdx.x < o) r[threadIdx.x] += r[threadIdx.x+o];
        __syncthreads();
    }
    if (threadIdx.x == 0) atomicAdd(&st[32], r[0]);
}

__global__ void k_absmax(const float* __restrict__ x, const float* __restrict__ lnw,
                         const float* __restrict__ lnb, float* __restrict__ st) {
    int s = blockIdx.x >> 6, chunk = blockIdx.x & 63;
    float mean = st[s] * (1.0f/CHW);
    float var  = st[16+s] * (1.0f/CHW) - mean*mean;
    float rstd = rsqrtf(var + GN_EPS);
    const float4* xs = (const float4*)(x + (size_t)s*CHW + (size_t)chunk*16384);
    float m = 0.f;
    for (int i = threadIdx.x; i < 4096; i += 256) {
        int c = (chunk*16384 + i*4) >> 12;
        float g = lnw[c]*rstd, b = lnb[c] - mean*g;
        float4 v = xs[i];
        m = fmaxf(m, fmaxf(fmaxf(fabsf(v.x*g+b), fabsf(v.y*g+b)),
                           fmaxf(fabsf(v.z*g+b), fabsf(v.w*g+b))));
    }
    __shared__ float r[256];
    r[threadIdx.x] = m; __syncthreads();
    for (int o = 128; o > 0; o >>= 1) {
        if (threadIdx.x < o) r[threadIdx.x] = fmaxf(r[threadIdx.x], r[threadIdx.x+o]);
        __syncthreads();
    }
    if (threadIdx.x == 0)
        atomicMax((unsigned int*)&st[33], __float_as_uint(r[0]));
}

__global__ void k_quantw(const float* __restrict__ w, const float* __restrict__ st,
                         signed char* __restrict__ wq) {
    float delta = 0.7f * st[32] * (1.0f/WELEMS);
    int i = blockIdx.x*256 + threadIdx.x;
    int co  = i / KTOT;
    int rem = i - co*KTOT;
    int ci  = rem / 9;
    int r   = rem - ci*9;
    float v = w[i];
    signed char q = (v > delta) ? (signed char)1 : ((v < -delta) ? (signed char)-1 : (signed char)0);
    wq[co*KTOT + r*256 + ci] = q;
}

__device__ __forceinline__ signed char q8(float v) {
    v = fminf(fmaxf(v, -QB + EPS), QB - EPS);
    float r = rintf(v);
    r = fminf(fmaxf(r, -128.f), 127.f);
    return (signed char)r;
}

__global__ void k_quantx(const float* __restrict__ x, const float* __restrict__ lnw,
                         const float* __restrict__ lnb, const float* __restrict__ st,
                         signed char* __restrict__ xq) {
    int base = (blockIdx.x*256 + threadIdx.x) * 4;
    int s = base / CHW;
    int e = base - s*CHW;
    int c = e >> 12;
    float mean = st[s] * (1.0f/CHW);
    float var  = st[16+s] * (1.0f/CHW) - mean*mean;
    float rstd = rsqrtf(var + GN_EPS);
    float gamma = fmaxf(st[33], EPS);
    float qs = QB / gamma;
    float g = lnw[c]*rstd*qs;
    float b = (lnb[c] - mean*lnw[c]*rstd)*qs;
    float4 v = *(const float4*)(x + base);
    char4 o;
    o.x = q8(v.x*g + b); o.y = q8(v.y*g + b); o.z = q8(v.z*g + b); o.w = q8(v.w*g + b);
    *(char4*)(xq + base) = o;
}


__global__ __launch_bounds__(256) void k_conv(const signed char* __restrict__ xq,
                                              const signed char* __restrict__ wq,
                                              const float* __restrict__ bias,
                                              const float* __restrict__ st,
                                              float* __restrict__ out) {
    __shared__ __align__(16) char smraw[128*68*4];
    signed char* Xs = (signed char*)smraw;
    signed char* Ws = (signed char*)(smraw + 64*80);
    float*       Ob = (float*)smraw;

    const int tid = threadIdx.x;
    const int smp = blockIdx.z;
    const int h   = blockIdx.y;
    const int co_base = blockIdx.x * 128;
    const int lane = tid & 31;
    const int wv = tid >> 5;
    const int wm = wv & 1, wn = wv >> 1;
    const int ml = lane & 15, hi = lane >> 4;

    v8i c00 = {0,0,0,0,0,0,0,0}, c01 = {0,0,0,0,0,0,0,0};
    v8i c10 = {0,0,0,0,0,0,0,0}, c11 = {0,0,0,0,0,0,0,0};

    const signed char* xs = xq + (size_t)smp * CHW;

    const int wcol = tid & 63;
    const int kseg = (tid >> 6) << 4;

    auto ldA = [&](int m) -> v8i {
        v8i a;
        const int* p0 = (const int*)(Xs + m*80 +      hi*16);
        const int* p1 = (const int*)(Xs + m*80 + 8 + hi*16);
        const int* p2 = (const int*)(Xs + m*80 + 32 + hi*16);
        const int* p3 = (const int*)(Xs + m*80 + 40 + hi*16);
        a[0]=p0[0]; a[1]=p0[1]; a[2]=p1[0]; a[3]=p1[1];
        a[4]=p2[0]; a[5]=p2[1]; a[6]=p3[0]; a[7]=p3[1];
        return a;
    };
    auto ldB = [&](int n) -> v8i {
        int4 lo = *(const int4*)(Ws + n*64 +      hi*16);
        int4 h4 = *(const int4*)(Ws + n*64 + 32 + hi*16);
        v8i b;
        b[0]=lo.x; b[1]=lo.y; b[2]=lo.z; b[3]=lo.w;
        b[4]=h4.x; b[5]=h4.y; b[6]=h4.z; b[7]=h4.w;
        return b;
    };

    for (int g = 0; g < 9; ++g) {
        int kh = g / 3, kw = g - (g/3)*3;
        int sh = h + kh - 1;
        int sw = wcol + kw - 1;
        bool ok = ((unsigned)sh < (unsigned)H) && ((unsigned)sw < (unsigned)W);
        const signed char* prow = xs + (sh << 6) + sw;

        #pragma unroll
        for (int sub = 0; sub < 4; ++sub) {
            const int k0 = g*256 + sub*64;
            {
                int co_l = tid >> 1, seg = tid & 1;
                const signed char* srcg = wq + (size_t)(co_base + co_l)*KTOT + k0 + seg*32;
                signed char* dstl = Ws + co_l*64 + seg*32;
#if USE_ASYNC
                __builtin_amdgcn_global_load_async_to_lds_b128(
                    (gv4p)srcg, (lv4p)dstl, 0, 0);
                __builtin_amdgcn_global_load_async_to_lds_b128(
                    (gv4p)(srcg + 16), (lv4p)(dstl + 16), 0, 0);
#else
                const int4* src = (const int4*)srcg;
                int4 a0 = src[0], a1 = src[1];
                int4* dst = (int4*)dstl;
                dst[0] = a0; dst[1] = a1;
#endif
            }
            {
                union { signed char c[16]; int4 v; } u;
                if (ok) {
                    const signed char* p = prow + (size_t)(sub*64 + kseg)*HW;
                    #pragma unroll
                    for (int j = 0; j < 16; j++) u.c[j] = p[(size_t)j*HW];
                } else {
                    u.v = make_int4(0, 0, 0, 0);
                }
                *(int4*)(Xs + wcol*80 + kseg) = u.v;
            }
#if USE_ASYNC
            wait_async0();
#endif
            __syncthreads();

            v8i a0f = ldA(wm*32 + ml);
            v8i a1f = ldA(wm*32 + 16 + ml);
            v8i b0f = ldB(wn*32 + ml);
            v8i b1f = ldB(wn*32 + 16 + ml);

            c00 = __builtin_amdgcn_wmma_i32_16x16x64_iu8(true, a0f, true, b0f, c00, false, false);
            c01 = __builtin_amdgcn_wmma_i32_16x16x64_iu8(true, a0f, true, b1f, c01, false, false);
            c10 = __builtin_amdgcn_wmma_i32_16x16x64_iu8(true, a1f, true, b0f, c10, false, false);
            c11 = __builtin_amdgcn_wmma_i32_16x16x64_iu8(true, a1f, true, b1f, c11, false, false);
            __syncthreads();
        }
    }

    float gamma = fmaxf(st[33], EPS);
    float dq = WSCALE * gamma * (1.0f/QB);
    auto dump = [&](v8i& acc, int Mb, int Nb) {
        int co = Nb + ml;
        int wb = Mb + 8*hi;
        float4 f0, f1;
        f0.x = (float)acc[0]*dq; f0.y = (float)acc[1]*dq;
        f0.z = (float)acc[2]*dq; f0.w = (float)acc[3]*dq;
        f1.x = (float)acc[4]*dq; f1.y = (float)acc[5]*dq;
        f1.z = (float)acc[6]*dq; f1.w = (float)acc[7]*dq;
        *(float4*)(Ob + co*68 + wb)     = f0;
        *(float4*)(Ob + co*68 + wb + 4) = f1;
    };
    dump(c00, wm*32,      wn*32);
    dump(c01, wm*32,      wn*32 + 16);
    dump(c10, wm*32 + 16, wn*32);
    dump(c11, wm*32 + 16, wn*32 + 16);
    __syncthreads();
    {
        int co_l = tid >> 1, seg = tid & 1;
        float bv = bias[co_base + co_l];
        float* dst = out + (((size_t)smp*C + co_base + co_l)*H + h)*W + seg*32;
        const float* src = Ob + co_l*68 + seg*32;
        #pragma unroll
        for (int i = 0; i < 8; i++) {
            float4 v = ((const float4*)src)[i];
            v.x += bv; v.y += bv; v.z += bv; v.w += bv;
            ((float4*)dst)[i] = v;
        }
    }
}


extern "C" void kernel_launch(void* const* d_in, const int* in_sizes, int n_in,
                              void* d_out, int out_size, void* d_ws, size_t ws_size,
                              hipStream_t stream) {
    (void)in_sizes; (void)n_in; (void)out_size; (void)ws_size;
    const float* x    = (const float*)d_in[0];
    const float* wgt  = (const float*)d_in[1];
    const float* bias = (const float*)d_in[2];
    const float* lnw  = (const float*)d_in[3];
    const float* lnb  = (const float*)d_in[4];
    float* out = (float*)d_out;

    char* ws = (char*)d_ws;
    signed char* xq = (signed char*)(ws + XQ_OFF);
    signed char* wq = (signed char*)(ws + WQ_OFF);
    float* st = (float*)(ws + ST_OFF);

    k_init   <<<1, 64, 0, stream>>>(st);
    k_gnstats<<<16*64, 256, 0, stream>>>(x, st);
    k_wabs   <<<576, 256, 0, stream>>>(wgt, st);
    k_absmax <<<16*64, 256, 0, stream>>>(x, lnw, lnb, st);
    k_quantw <<<WELEMS/256, 256, 0, stream>>>(wgt, st, wq);
    k_quantx <<<16384, 256, 0, stream>>>(x, lnw, lnb, st, xq);
    dim3 g(2, 64, 16);
    k_conv   <<<g, 256, 0, stream>>>(xq, wq, bias, st, out);
}
